// SimpleLSTM_4_61452392071769
// MI455X (gfx1250) — matrix operand order repaired and checked by reading, not compiled, not run on hardware
//
#include <hip/hip_runtime.h>
#include <hip/hip_bf16.h>
#include <math.h>

// Problem constants (match reference)
#define B_TOTAL 4096
#define T_STEPS 512
#define IN_SZ   20
#define HID     51
#define GSTRIDE 52      // padded per-gate column stride (51 -> 52)
#define NG      208     // 4 * 52 padded gate columns
#define NTILES  13      // 208 / 16 N-tiles
#define KH      64      // padded hidden K dim (51 -> 64)
#define KX      32      // padded input K dim (20 -> 32)
#define HROW    72      // sH row stride in halves (bank-conflict pad)
#define GROW    20      // sG row stride in floats (bank-conflict pad)

#define THREADS  128    // 4 wave32s cooperating on one 16-row batch tile
#define NWAVES   4
#define EW_TOTAL (16 * GSTRIDE)   // 832 gate elements
#define EW_FULL  (EW_TOTAL / THREADS)   // 6 full sweeps; tail covers waves 0..1

typedef _Float16 v16h __attribute__((ext_vector_type(16)));
typedef _Float16 v8h  __attribute__((ext_vector_type(8)));
typedef float    v8f  __attribute__((ext_vector_type(8)));
typedef float    v4f  __attribute__((ext_vector_type(4)));

#if __has_builtin(__builtin_amdgcn_tanhf)
__device__ __forceinline__ float fast_tanh(float x) { return __builtin_amdgcn_tanhf(x); }
#elif __has_builtin(__builtin_amdgcn_tanh_f32)
__device__ __forceinline__ float fast_tanh(float x) { return __builtin_amdgcn_tanh_f32(x); }
#else
__device__ __forceinline__ float fast_tanh(float x) { return tanhf(x); }
#endif

__device__ __forceinline__ float sigmoidf_(float x) {
  return 0.5f * fast_tanh(0.5f * x) + 0.5f;
}

__device__ __forceinline__ v8f zero8f() {
  v8f z;
#pragma unroll
  for (int i = 0; i < 8; ++i) z[i] = 0.f;
  return z;
}

__device__ __forceinline__ v16h zero16h() {
  v16h z;
#pragma unroll
  for (int i = 0; i < 16; ++i) z[i] = (_Float16)0.f;
  return z;
}

__device__ __forceinline__ v16h cat8(v8h a, v8h b) {
  v16h r;
#pragma unroll
  for (int i = 0; i < 8; ++i) { r[i] = a[i]; r[i + 8] = b[i]; }
  return r;
}

__device__ __forceinline__ v8f wmma_f16(v16h a, v16h b, v8f c) {
  // D = A(16x32 f16) * B(32x16 f16) + C(16x16 f32)
  return __builtin_amdgcn_wmma_f32_16x16x32_f16(false, a, false, b, (short)0, c,
                                                false, false);
}

__global__ __launch_bounds__(THREADS)
void lstm2_wmma_kernel(const float* __restrict__ x,
                       const float* __restrict__ Wih1, const float* __restrict__ Whh1,
                       const float* __restrict__ bih1, const float* __restrict__ bhh1,
                       const float* __restrict__ Wih2, const float* __restrict__ Whh2,
                       const float* __restrict__ bih2, const float* __restrict__ bhh2,
                       const float* __restrict__ Wmu,  const float* __restrict__ bmu,
                       const float* __restrict__ Wlv,  const float* __restrict__ blv,
                       float* __restrict__ out)
{
  // LDS: 26624 + 13312 + 832 + 16640 + 2304 = 59712 bytes (shared by 4 waves)
  __shared__ __align__(64) _Float16 sWhh[NG * KH];   // B-frag layout: [n][k] row-major f16
  __shared__ __align__(64) _Float16 sWih[NG * KX];
  __shared__ __align__(64) float    sB[NG];          // combined bias, padded layout
  __shared__ __align__(64) float    sG[NG * GROW];   // gate dump: [col][row], stride 20
  __shared__ __align__(64) _Float16 sH[16 * HROW];   // h1 staging, f16, stride 72

  const int tid  = (int)threadIdx.x;   // 0..127
  const int lane = tid & 31;
  // Wave id is identical across all 32 lanes: pin it to an SGPR so all loop
  // control / work-split branches are scalar (SALU) and EXEC stays all-1s
  // around every WMMA.
  const int wid  = __builtin_amdgcn_readfirstlane(tid >> 5);   // 0..3
  const int mrow = lane & 15;          // batch row in tile / column in N-tile
  const int hi   = (lane >> 4) & 1;    // lane-group select
  const int b0   = (int)blockIdx.x * 16;

  // ---------------- one-time staging (all 128 threads) ----------------
  for (int idx = tid; idx < NG * KH; idx += THREADS) {
    int n = idx >> 6, k = idx & 63;
    int g = n / GSTRIDE, j = n - g * GSTRIDE;
    float w = (j < HID && k < HID) ? Whh1[(g * HID + j) * HID + k] : 0.f;
    sWhh[idx] = (_Float16)w;
  }
  for (int idx = tid; idx < NG * KX; idx += THREADS) {
    int n = idx >> 5, k = idx & 31;
    int g = n / GSTRIDE, j = n - g * GSTRIDE;
    float w = (j < HID && k < IN_SZ) ? Wih1[(g * HID + j) * IN_SZ + k] : 0.f;
    sWih[idx] = (_Float16)w;
  }
  for (int n = tid; n < NG; n += THREADS) {
    int g = n / GSTRIDE, j = n - g * GSTRIDE;
    sB[n] = (j < HID) ? (bih1[g * HID + j] + bhh1[g * HID + j]) : 0.f;
  }
  for (int idx = tid; idx < 16 * HROW; idx += THREADS) sH[idx] = (_Float16)0.f;

  // Layer-2 B fragments (Wih2 padded to 64x16; only wave 0's copy is used)
  v16h w2f0 = zero16h(), w2f1 = zero16h();
  {
    const int koff = hi ? 16 : 0;
#pragma unroll
    for (int i = 0; i < 16; ++i) {
      int k0 = koff + i;
      int k1 = 32 + koff + i;
      if (mrow < 4) {
        if (k0 < HID) w2f0[i] = (_Float16)Wih2[mrow * HID + k0];
        if (k1 < HID) w2f1[i] = (_Float16)Wih2[mrow * HID + k1];
      }
    }
  }
  // tiny layer-2 / head scalars (uniform across lanes)
  const float bq0 = bih2[0] + bhh2[0], bq1 = bih2[1] + bhh2[1];
  const float bq2 = bih2[2] + bhh2[2], bq3 = bih2[3] + bhh2[3];
  const float wh0 = Whh2[0], wh1 = Whh2[1], wh2 = Whh2[2], wh3 = Whh2[3];
  const float wmu = Wmu[0], vbmu = bmu[0], wlv = Wlv[0], vblv = blv[0];

  __syncthreads();

  // ---------------- recurrent state ----------------
  v16h ah0 = zero16h(), ah1 = zero16h();   // h1 A-fragments (K-tiles 0,1)
  float c1[EW_FULL + 1];
#pragma unroll
  for (int i = 0; i <= EW_FULL; ++i) c1[i] = 0.f;
  float h2 = 0.f, c2 = 0.f, h2sum = 0.f;   // meaningful on wave 0 only

  const float* xbase = x + (size_t)(b0 + mrow) * (size_t)T_STEPS * IN_SZ;

  for (int t = 0; t < T_STEPS; ++t) {
    // ---- build A_x fragment (16x32 f16, K 20..31 zero); shared x tile hits WGP$ ----
    const float* xr = xbase + (size_t)t * IN_SZ;
    __builtin_prefetch(xr + 8 * IN_SZ, 0, 0);     // stream upcoming timesteps
    float4 p0 = *(const float4*)(xr + (hi ? 8 : 0));
    float4 p1 = *(const float4*)(xr + (hi ? 12 : 4));
    float4 p2; p2.x = p2.y = p2.z = p2.w = 0.f;
    if (!hi) p2 = *(const float4*)(xr + 16);      // K 16..19 (rest of pad = 0)
    v16h ax = zero16h();
    ax[0] = (_Float16)p0.x; ax[1] = (_Float16)p0.y; ax[2] = (_Float16)p0.z; ax[3] = (_Float16)p0.w;
    ax[4] = (_Float16)p1.x; ax[5] = (_Float16)p1.y; ax[6] = (_Float16)p1.z; ax[7] = (_Float16)p1.w;
    ax[8] = (_Float16)p2.x; ax[9] = (_Float16)p2.y; ax[10] = (_Float16)p2.z; ax[11] = (_Float16)p2.w;

    // ---- layer-1 gates: N-tiles strided across the 4 waves (scalar loop) ----
#pragma unroll 1
    for (int nt = wid; nt < NTILES; nt += NWAVES) {
      const int n = nt * 16 + mrow;
      const _Float16* pbi = &sWih[n * KX + (hi ? 8 : 0)];
      const _Float16* pbh = &sWhh[n * KH + (hi ? 16 : 0)];
      v16h bih = cat8(*(const v8h*)pbi, *(const v8h*)(pbi + 16));
      v16h bh0 = *(const v16h*)pbh;
      v16h bh1 = *(const v16h*)(pbh + 32);
      v8f acc = wmma_f16(ax, bih, zero8f());
      acc = wmma_f16(ah0, bh0, acc);
      acc = wmma_f16(ah1, bh1, acc);
      // dump D tile -> sG[col][row]; per-lane 2x contiguous b128 stores
      float* gp = &sG[n * GROW + hi * 8];
      *(v4f*)gp       = __builtin_shufflevector(acc, acc, 0, 1, 2, 3);
      *(v4f*)(gp + 4) = __builtin_shufflevector(acc, acc, 4, 5, 6, 7);
    }

    __syncthreads();   // all gate tiles dumped

    // ---- elementwise LSTM cell 1: 832 elements / 128 threads ----
    // 6 full sweeps, plus a tail that is live exactly for waves 0..1 (scalar branch).
#pragma unroll
    for (int i = 0; i <= EW_FULL; ++i) {
      if (i == EW_FULL && wid >= 2) break;   // uniform per wave
      const int e = tid + THREADS * i;       // < 832 by construction
      const int j = e >> 4;                  // hidden col 0..51 (51 = pad, stays 0)
      const int m = e & 15;                  // batch row
      const int a0 = j * GROW + m;
      float gi = sG[a0]                     + sB[j];
      float gf = sG[a0 +     GSTRIDE*GROW]  + sB[    GSTRIDE + j];
      float gg = sG[a0 + 2 * GSTRIDE*GROW]  + sB[2 * GSTRIDE + j];
      float go = sG[a0 + 3 * GSTRIDE*GROW]  + sB[3 * GSTRIDE + j];
      float cc = sigmoidf_(gf) * c1[i] + sigmoidf_(gi) * fast_tanh(gg);
      c1[i] = cc;
      float hh = sigmoidf_(go) * fast_tanh(cc);
      sH[m * HROW + j] = (_Float16)hh;
    }

    __syncthreads();   // h1 tile staged for everyone

    // ---- rebuild h1 A-fragments (serve layer-2 now and layer-1 next step) ----
    {
      const _Float16* p = &sH[mrow * HROW + (hi ? 16 : 0)];
      ah0 = cat8(*(const v8h*)p,        *(const v8h*)(p + 8));
      ah1 = cat8(*(const v8h*)(p + 32), *(const v8h*)(p + 40));
    }

    // ---- layer-2 gates via WMMA on wave 0 only (scalar branch; owns nt0 cols) ----
    if (wid == 0) {
      v8f a2 = wmma_f16(ah0, w2f0, zero8f());
      a2 = wmma_f16(ah1, w2f1, a2);
      float* gp2 = &sG[mrow * GROW + hi * 8];
      *(v4f*)gp2       = __builtin_shufflevector(a2, a2, 0, 1, 2, 3);
      *(v4f*)(gp2 + 4) = __builtin_shufflevector(a2, a2, 4, 5, 6, 7);

      float g0 = sG[0 * GROW + mrow] + bq0 + wh0 * h2;  // i
      float g1 = sG[1 * GROW + mrow] + bq1 + wh1 * h2;  // f
      float g2 = sG[2 * GROW + mrow] + bq2 + wh2 * h2;  // g
      float g3 = sG[3 * GROW + mrow] + bq3 + wh3 * h2;  // o
      float c2n = sigmoidf_(g1) * c2 + sigmoidf_(g0) * fast_tanh(g2);
      c2 = c2n;
      h2 = sigmoidf_(g3) * fast_tanh(c2n);
      h2sum += h2;
    }
  }

  // ---------------- head: mean over T, mu / log_var / CI ----------------
  if (wid == 0) {
    if (lane < 16) {
      const int b = b0 + mrow;
      float agg = h2sum * (1.0f / (float)T_STEPS);
      float mu = wmu * agg + vbmu;
      float lv = wlv * agg + vblv;
      float sg = __expf(0.5f * lv);
      out[b]               = mu - 1.96f * sg;  // lower
      out[B_TOTAL + b]     = mu;               // mu
      out[2 * B_TOTAL + b] = mu + 1.96f * sg;  // upper
      out[3 * B_TOTAL + b] = lv;               // log_var
    }
  }
}

extern "C" void kernel_launch(void* const* d_in, const int* in_sizes, int n_in,
                              void* d_out, int out_size, void* d_ws, size_t ws_size,
                              hipStream_t stream) {
  (void)in_sizes; (void)n_in; (void)out_size; (void)d_ws; (void)ws_size;
  const float* x    = (const float*)d_in[0];
  const float* Wih1 = (const float*)d_in[1];
  const float* Whh1 = (const float*)d_in[2];
  const float* bih1 = (const float*)d_in[3];
  const float* bhh1 = (const float*)d_in[4];
  const float* Wih2 = (const float*)d_in[5];
  const float* Whh2 = (const float*)d_in[6];
  const float* bih2 = (const float*)d_in[7];
  const float* bhh2 = (const float*)d_in[8];
  const float* Wmu  = (const float*)d_in[9];
  const float* bmu  = (const float*)d_in[10];
  const float* Wlv  = (const float*)d_in[11];
  const float* blv  = (const float*)d_in[12];
  float* out = (float*)d_out;

  dim3 grid(B_TOTAL / 16), block(THREADS);
  hipLaunchKernelGGL(lstm2_wmma_kernel, grid, block, 0, stream,
                     x, Wih1, Whh1, bih1, bhh1, Wih2, Whh2, bih2, bhh2,
                     Wmu, bmu, Wlv, blv, out);
}
